// IBRNet_31705448579528
// MI455X (gfx1250) — matrix operand order repaired and checked by reading, not compiled, not run on hardware
//
#include <hip/hip_runtime.h>

typedef __attribute__((ext_vector_type(16))) _Float16 v16h;
typedef __attribute__((ext_vector_type(8)))  float    v8f;

#define N_SAMP  64
#define N_VIEW  10
#define CH_SAMP 16
#define CH_ROWS 160
#define NCHUNK  4
#define BLOCK   256
#define NWAVE   8

// ---- packed weight blob (d_ws) layout --------------------------------------
// f16 region: transposed, zero-padded weights Wt[n][k] (K padded to mult of 32)
enum {
  H_RD1T = 0,      // [32][32]   rd_w1  (25,32)
  H_RD2T = 1024,   // [48][32]   rd_w2  (32,35)
  H_BASE1T = 2560, // [64][128]  base_w1(105,64)
  H_BASE2T = 10752,// [32][64]   base_w2(64,32)
  H_VIS1T = 12800, // [32][32]   vis_w1
  H_VIS2T = 13824, // [48][32]   vis_w2 (32,33)
  H_V21T  = 15360, // [32][32]   vis2_w1
  H_RD21T = 16384, // [32][32]   rd2_w1
  H_RD22T = 17408, // [32][32]   rd2_w2 (32,24)
  H_GEO1T = 18432, // [64][96]   geo_w1 (65,64)
  H_GEO2T = 24576, // [16][64]   geo_w2 (64,16)
  H_RGB1T = 25600, // [16][64]   rgb_w1 (58,16)
  H_RGB2T = 26624, // [16][32]   rgb_w2 (16,8)
  H_END   = 27136
};
#define WS_HALF_BYTES (H_END * 2)
// f32 region (after f16 region): biases (zero padded) + small weights
enum {
  F_RD_B1=0, F_RD_B2=32, F_BASE_B1=80, F_BASE_B2=144,
  F_VIS_B1=176, F_VIS_B2=208, F_V2_B1=256, F_V2_W2=288, F_V2_B2=320,
  F_RD2_B1=324, F_RD2_B2=356, F_GEO_B1=388, F_GEO_B2=452,
  F_RGB_B1=468, F_RGB_B2=484, F_RGB_W3=500, F_RGB_B3=508,
  F_OG_W1=512, F_OG_B1=768, F_OG_W2=784, F_OG_B2=800,
  F_WQI=804, F_WKI=1060, F_WVI=1316, F_FC1=1572,
  F_WQMT=1828, F_WKMT=2020, F_LFW0=2212, F_LFW1=2596,
  F_BETA0=3364, F_BETA1=3556, F_LNG=3604, F_LNB=3620, F_FEND=3636
};

// ---- LDS layout (dynamic) --------------------------------------------------
#define OFF_WH     0        // f16 weight blob copy          54272 B
#define OFF_SA     54272    // f16 [160][128]                40960
#define OFF_SB     95232    // f16 [160][64]                 20480
#define OFF_SC     115712   // f16 [160][64]                 20480
#define OFF_SFEAT  136192   // f16 [160][48]                 15360
#define OFF_SX     151552   // f16 [160][32]                 10240
#define OFF_SRD    161792   // f16 [160][32]                 10240
#define OFF_SMASK  172032   // f32 [160]
#define OFF_SWGT   172672   // f32 [160]
#define OFF_SVIS   173312   // f32 [160]
#define OFF_SRGBIN 173952   // f32 [160][3]
#define OFF_GFIN   175872   // f16 [64][96]                  12288
#define OFF_GF     188160   // f32 [64][16]  (inv = gf + PE)
#define OFF_LFT    192256   // f32 [64][48]  lf_tensor
#define OFF_NV     204544   // f32 [64]      num_valid
#define OFF_RGBO   204800   // f32 [64][4]
#define OFF_ST     205824   // f32 [64][16]  attention out (post-LN)
#define LDS_BYTES  209920
// attention-phase overlays on SA..SC region:
#define OFF_QF   (OFF_SA)            // f32 [64][4][28] = 28672
#define OFF_KF   (OFF_SA + 28672)
#define OFF_VF   (OFF_SA + 57344)    // f32 [64][16]
#define OFF_AOUT (OFF_SA + 61440)    // f32 [64][16]

struct PP { const float* p[49]; };

// fast hardware transcendentals (v_exp_f32 path, branchless) --------------
__device__ __forceinline__ float fexp(float x)  { return __expf(x); }
__device__ __forceinline__ float eluf(float x)  { return x > 0.f ? x : (__expf(x) - 1.f); }
__device__ __forceinline__ float sigm(float x)  { return 1.f / (1.f + __expf(-x)); }

// C[16x16 tile] = ELU(A[M x K] * Wt^T + bias) (+res), all f16 LDS operands,
// f32 WMMA accumulation via v_wmma_f32_16x16x32_f16.
__device__ __forceinline__ void gemm_lds(
    const _Float16* __restrict__ A, int lda,
    const _Float16* __restrict__ Bt, int ldb,
    _Float16* __restrict__ D, int ldd,
    const float* __restrict__ bias,
    const _Float16* __restrict__ res, int ldr,
    int Mtiles, int Ntiles, int K, int tid)
{
  const int wave = tid >> 5;
  const int lane = tid & 31;
  const int lr = lane & 15;
  const int hf = lane >> 4;
  const int nt = Mtiles * Ntiles;
  for (int t = wave; t < nt; t += NWAVE) {
    const int mB = (t / Ntiles) << 4;
    const int nB = (t % Ntiles) << 4;
    const _Float16* Ar = A + (mB + lr) * lda;
    const _Float16* Bc = Bt + (nB + lr) * ldb;
    v8f acc = {};
    for (int k0 = 0; k0 < K; k0 += 32) {
      v16h av, bv;
#pragma unroll
      for (int j = 0; j < 8; ++j) {
        // 16-bit A 16x32 layout: V0..3 hold K{0..7 | +8 by lane-half}, V4..7 +16
        const int ka = k0 + ((j & 3) << 1) + ((j >> 2) << 4) + (hf << 3);
        av[2 * j]     = Ar[ka];
        av[2 * j + 1] = Ar[ka + 1];
        // 16-bit B 32x16 layout: lane=col, same K order as A: K{0..7 | +8 by lane-half}, then +16
        const int kb = k0 + ((j & 3) << 1) + ((j >> 2) << 4) + (hf << 3);
        bv[2 * j]     = Bc[kb];
        bv[2 * j + 1] = Bc[kb + 1];
      }
      acc = __builtin_amdgcn_wmma_f32_16x16x32_f16(false, av, false, bv,
                                                   (short)0, acc, false, false);
    }
    const int col = nB + lr;
    const float bs = bias[col];
#pragma unroll
    for (int j = 0; j < 8; ++j) {
      const int row = mB + j + hf * 8;  // C/D layout: VGPR j -> M=j(+8 for hi half)
      float v = eluf(acc[j] + bs);
      if (res) v += (float)res[row * ldr + col];
      D[row * ldd + col] = (_Float16)v;
    }
  }
}

// ---- weight packing kernel -------------------------------------------------
__global__ __launch_bounds__(BLOCK) void ibrnet_prep_kernel(PP P, _Float16* WH, float* FW) {
  const int tid = threadIdx.x;
  auto putT = [&](int off, const float* W, int K0, int N0, int Kp, int Np) {
    for (int i = tid; i < Np * Kp; i += BLOCK) {
      int n = i / Kp, k = i - n * Kp;
      WH[off + i] = (_Float16)((k < K0 && n < N0) ? W[k * N0 + n] : 0.f);
    }
  };
  auto putF = [&](int off, const float* B, int N0, int Np) {
    for (int i = tid; i < Np; i += BLOCK) FW[off + i] = (i < N0) ? B[i] : 0.f;
  };
  putT(H_RD1T,  P.p[28], 25, 32, 32, 32);
  putT(H_RD2T,  P.p[29], 32, 35, 32, 48);
  putT(H_BASE1T,P.p[5], 105, 64,128, 64);
  putT(H_BASE2T,P.p[6],  64, 32, 64, 32);
  putT(H_VIS1T, P.p[42], 32, 32, 32, 32);
  putT(H_VIS2T, P.p[43], 32, 33, 32, 48);
  putT(H_V21T,  P.p[38], 32, 32, 32, 32);
  putT(H_RD21T, P.p[24], 32, 32, 32, 32);
  putT(H_RD22T, P.p[25], 32, 24, 32, 32);
  putT(H_GEO1T, P.p[10], 65, 64, 96, 64);
  putT(H_GEO2T, P.p[11], 64, 16, 64, 16);
  putT(H_RGB1T, P.p[33], 58, 16, 64, 16);
  putT(H_RGB2T, P.p[34], 16,  8, 32, 16);
  putF(F_RD_B1,  P.p[26], 32, 32);   putF(F_RD_B2,  P.p[27], 35, 48);
  putF(F_BASE_B1,P.p[3],  64, 64);   putF(F_BASE_B2,P.p[4],  32, 32);
  putF(F_VIS_B1, P.p[40], 32, 32);   putF(F_VIS_B2, P.p[41], 33, 48);
  putF(F_V2_B1,  P.p[36], 32, 32);   putF(F_V2_W2,  P.p[39], 32, 32);
  putF(F_V2_B2,  P.p[37],  1,  4);
  putF(F_RD2_B1, P.p[22], 32, 32);   putF(F_RD2_B2, P.p[23], 24, 32);
  putF(F_GEO_B1, P.p[8],  64, 64);   putF(F_GEO_B2, P.p[9],  16, 16);
  putF(F_RGB_B1, P.p[30], 16, 16);   putF(F_RGB_B2, P.p[31],  8, 16);
  putF(F_RGB_W3, P.p[35],  8,  8);   putF(F_RGB_B3, P.p[32],  1,  4);
  putF(F_OG_W1,  P.p[20], 256, 256); putF(F_OG_B1,  P.p[18], 16, 16);
  putF(F_OG_W2,  P.p[21], 16, 16);   putF(F_OG_B2,  P.p[19],  1,  4);
  putF(F_WQI,    P.p[46], 256, 256); putF(F_WKI,    P.p[44], 256, 256);
  putF(F_WVI,    P.p[48], 256, 256); putF(F_FC1,    P.p[7],  256, 256);
  putF(F_WQMT,   P.p[47], 192, 192); putF(F_WKMT,   P.p[45], 192, 192);
  putF(F_LFW0,   P.p[14], 384, 384); putF(F_LFW1,   P.p[15], 768, 768);
  putF(F_BETA0,  P.p[12], 192, 192); putF(F_BETA1,  P.p[13], 48, 48);
  putF(F_LNG,    P.p[17], 16, 16);   putF(F_LNB,    P.p[16], 16, 16);
}

// ---- main kernel: one workgroup per ray ------------------------------------
__global__ __launch_bounds__(BLOCK)
void ibrnet_ray_kernel(const float* __restrict__ rgb_feat,
                       const float* __restrict__ ray_diff,
                       const float* __restrict__ mask,
                       const _Float16* __restrict__ WHg,
                       const float* __restrict__ FW,
                       float* __restrict__ out)
{
  extern __shared__ char smem[];
  const int tid = threadIdx.x;
  const int ray = blockIdx.x;

  _Float16* WH    = (_Float16*)(smem + OFF_WH);
  _Float16* sA    = (_Float16*)(smem + OFF_SA);
  _Float16* sB    = (_Float16*)(smem + OFF_SB);
  _Float16* sC    = (_Float16*)(smem + OFF_SC);
  _Float16* sFeat = (_Float16*)(smem + OFF_SFEAT);
  _Float16* sX    = (_Float16*)(smem + OFF_SX);
  _Float16* sRD   = (_Float16*)(smem + OFF_SRD);
  float* sMask  = (float*)(smem + OFF_SMASK);
  float* sWgt   = (float*)(smem + OFF_SWGT);
  float* sVis   = (float*)(smem + OFF_SVIS);
  float* sRGBin = (float*)(smem + OFF_SRGBIN);
  _Float16* sGFIN = (_Float16*)(smem + OFF_GFIN);
  float* sGF   = (float*)(smem + OFF_GF);
  float* sLFT  = (float*)(smem + OFF_LFT);
  float* sNV   = (float*)(smem + OFF_NV);
  float* sRGBO = (float*)(smem + OFF_RGBO);
  float* sT    = (float*)(smem + OFF_ST);

  // copy packed f16 weights into LDS
  {
    const uint32_t* s = (const uint32_t*)WHg;
    uint32_t* d = (uint32_t*)WH;
    for (int i = tid; i < H_END / 2; i += BLOCK) d[i] = s[i];
  }
  __syncthreads();

  for (int ch = 0; ch < NCHUNK; ++ch) {
    const int s0 = ch * CH_SAMP;
    const int rowBase = ray * (N_SAMP * N_VIEW) + s0 * N_VIEW;
    if (ch + 1 < NCHUNK) {  // pull next chunk toward cache
      const float* nx = rgb_feat + (rowBase + CH_ROWS) * 35;
      __builtin_prefetch(nx + tid * 16, 0, 1);
    }
    // ---- loads (f32 -> f16, zero padded) ----
    for (int i = tid; i < CH_ROWS * 32; i += BLOCK) {
      int r = i >> 5, c = i & 31;
      sRD[i] = (_Float16)((c < 25) ? ray_diff[(rowBase + r) * 25 + c] : 0.f);
    }
    for (int i = tid; i < CH_ROWS * 48; i += BLOCK) {
      int r = i / 48, c = i - r * 48;
      sFeat[i] = (_Float16)((c < 35) ? rgb_feat[(rowBase + r) * 35 + c] : 0.f);
    }
    for (int i = tid; i < CH_ROWS * 3; i += BLOCK) {
      int r = i / 3, c = i - r * 3;
      sRGBin[i] = rgb_feat[(rowBase + r) * 35 + c];
    }
    for (int i = tid; i < CH_ROWS; i += BLOCK) sMask[i] = mask[rowBase + i];
    __syncthreads();
    // ---- weight1 = mask / (sum_views + eps); num_valid ----
    for (int smp = tid; smp < CH_SAMP; smp += BLOCK) {
      float su = 0.f;
      for (int v = 0; v < N_VIEW; ++v) su += sMask[smp * N_VIEW + v];
      sNV[s0 + smp] = su;
      float inv = 1.f / (su + 1e-8f);
      for (int v = 0; v < N_VIEW; ++v) sWgt[smp * N_VIEW + v] = sMask[smp * N_VIEW + v] * inv;
    }
    __syncthreads();
    // ---- dfeat MLP (25->32->35) + residual into feat ----
    gemm_lds(sRD, 32, WH + H_RD1T, 32, sB, 32, FW + F_RD_B1, nullptr, 0, 10, 2, 32, tid);
    __syncthreads();
    gemm_lds(sB, 32, WH + H_RD2T, 32, sFeat, 48, FW + F_RD_B2, sFeat, 48, 10, 3, 32, tid);
    __syncthreads();
    // ---- weighted mean/var of feat; build x_in = [mean,var,feat] (105 pad 128) ----
    for (int i = tid; i < CH_SAMP * 35; i += BLOCK) {
      int smp = i / 35, c = i - smp * 35;
      float mean = 0.f, var = 0.f;
      for (int v = 0; v < N_VIEW; ++v)
        mean += sWgt[smp * N_VIEW + v] * (float)sFeat[(smp * N_VIEW + v) * 48 + c];
      for (int v = 0; v < N_VIEW; ++v) {
        float d = (float)sFeat[(smp * N_VIEW + v) * 48 + c] - mean;
        var += sWgt[smp * N_VIEW + v] * d * d;
      }
      _Float16 hm = (_Float16)mean, hv = (_Float16)var;
      for (int v = 0; v < N_VIEW; ++v) {
        int r = smp * N_VIEW + v;
        sA[r * 128 + c] = hm;
        sA[r * 128 + 35 + c] = hv;
        sA[r * 128 + 70 + c] = sFeat[r * 48 + c];
      }
    }
    for (int i = tid; i < CH_ROWS * 23; i += BLOCK) {
      int r = i / 23, c = 105 + (i - r * 23);
      sA[r * 128 + c] = (_Float16)0.f;
    }
    __syncthreads();
    // ---- base MLP (105->64->32) ----
    gemm_lds(sA, 128, WH + H_BASE1T, 128, sB, 64, FW + F_BASE_B1, nullptr, 0, 10, 4, 128, tid);
    __syncthreads();
    gemm_lds(sB, 64, WH + H_BASE2T, 64, sC, 32, FW + F_BASE_B2, nullptr, 0, 10, 2, 64, tid);
    __syncthreads();
    // ---- vis MLP on x*weight ----
    for (int i = tid; i < CH_ROWS * 32; i += BLOCK) {
      int r = i >> 5;
      sB[i] = (_Float16)((float)sC[i] * sWgt[r]);
    }
    __syncthreads();
    gemm_lds(sB, 32, WH + H_VIS1T, 32, sA, 32, FW + F_VIS_B1, nullptr, 0, 10, 2, 32, tid);
    __syncthreads();
    gemm_lds(sA, 32, WH + H_VIS2T, 32, sB, 48, FW + F_VIS_B2, nullptr, 0, 10, 3, 32, tid);
    __syncthreads();
    // vis = sigmoid(col32)*mask ; x = x0 + x_res
    for (int i = tid; i < CH_ROWS; i += BLOCK)
      sVis[i] = sigm((float)sB[i * 48 + 32]) * sMask[i];
    for (int i = tid; i < CH_ROWS * 32; i += BLOCK) {
      int r = i >> 5, c = i & 31;
      sX[i] = (_Float16)((float)sC[i] + (float)sB[r * 48 + c]);
    }
    __syncthreads();
    // ---- vis2 MLP on x*vis -> new vis, weight2 ----
    for (int i = tid; i < CH_ROWS * 32; i += BLOCK) {
      int r = i >> 5;
      sA[i] = (_Float16)((float)sX[i] * sVis[r]);
    }
    __syncthreads();
    gemm_lds(sA, 32, WH + H_V21T, 32, sB, 32, FW + F_V2_B1, nullptr, 0, 10, 2, 32, tid);
    __syncthreads();
    for (int i = tid; i < CH_ROWS; i += BLOCK) {
      float acc = FW[F_V2_B2];
#pragma unroll
      for (int c = 0; c < 32; ++c) acc += (float)sB[i * 32 + c] * FW[F_V2_W2 + c];
      sVis[i] = sigm(acc) * sMask[i];
    }
    __syncthreads();
    for (int smp = tid; smp < CH_SAMP; smp += BLOCK) {
      float su = 0.f;
      for (int v = 0; v < N_VIEW; ++v) su += sVis[smp * N_VIEW + v];
      float inv = 1.f / (su + 1e-8f);
      float wm = 0.f;
      for (int v = 0; v < N_VIEW; ++v) {
        float w = sVis[smp * N_VIEW + v] * inv;
        sWgt[smp * N_VIEW + v] = w;
        wm += w;
      }
      sGFIN[(s0 + smp) * 96 + 64] = (_Float16)(wm * 0.1f);
      for (int c = 65; c < 96; ++c) sGFIN[(s0 + smp) * 96 + c] = (_Float16)0.f;
    }
    __syncthreads();
    // ---- geo input: weighted mean/var of x ----
    for (int i = tid; i < CH_SAMP * 32; i += BLOCK) {
      int smp = i >> 5, c = i & 31;
      float mean = 0.f, var = 0.f;
      for (int v = 0; v < N_VIEW; ++v)
        mean += sWgt[smp * N_VIEW + v] * (float)sX[(smp * N_VIEW + v) * 32 + c];
      for (int v = 0; v < N_VIEW; ++v) {
        float d = (float)sX[(smp * N_VIEW + v) * 32 + c] - mean;
        var += sWgt[smp * N_VIEW + v] * d * d;
      }
      sGFIN[(s0 + smp) * 96 + c] = (_Float16)mean;
      sGFIN[(s0 + smp) * 96 + 32 + c] = (_Float16)var;
    }
    __syncthreads();
    // ---- lf = rd2 MLP (32->32->24); weighted stats -> lf_tensor ----
    gemm_lds(sX, 32, WH + H_RD21T, 32, sA, 32, FW + F_RD2_B1, nullptr, 0, 10, 2, 32, tid);
    __syncthreads();
    gemm_lds(sA, 32, WH + H_RD22T, 32, sB, 32, FW + F_RD2_B2, nullptr, 0, 10, 2, 32, tid);
    __syncthreads();
    for (int i = tid; i < CH_SAMP * 24; i += BLOCK) {
      int smp = i / 24, j = i - smp * 24, f = j >> 1, comp = j & 1;
      float mean = 0.f, var = 0.f;
      for (int v = 0; v < N_VIEW; ++v)
        mean += sWgt[smp * N_VIEW + v] * (float)sB[(smp * N_VIEW + v) * 32 + j];
      for (int v = 0; v < N_VIEW; ++v) {
        float d = (float)sB[(smp * N_VIEW + v) * 32 + j] - mean;
        var += sWgt[smp * N_VIEW + v] * d * d;
      }
      sLFT[(s0 + smp) * 48 + f * 4 + comp] = mean;
      sLFT[(s0 + smp) * 48 + f * 4 + 2 + comp] = var;
    }
    // ---- xr = [x, vis, ray_diff] (58 pad 64) ----
    for (int i = tid; i < CH_ROWS * 64; i += BLOCK) {
      int r = i >> 6, c = i & 63;
      float v;
      if (c < 32) v = (float)sX[r * 32 + c];
      else if (c == 32) v = sVis[r];
      else if (c < 58) v = (float)sRD[r * 32 + (c - 33)];
      else v = 0.f;
      sA[i] = (_Float16)v;
    }
    __syncthreads();
    gemm_lds(sA, 64, WH + H_RGB1T, 64, sB, 32, FW + F_RGB_B1, nullptr, 0, 10, 1, 64, tid);
    __syncthreads();
    for (int i = tid; i < CH_ROWS * 16; i += BLOCK) {
      int r = i >> 4, c = 16 + (i & 15);
      sB[r * 32 + c] = (_Float16)0.f;
    }
    __syncthreads();
    gemm_lds(sB, 32, WH + H_RGB2T, 32, sC, 16, FW + F_RGB_B2, nullptr, 0, 10, 1, 32, tid);
    __syncthreads();
    // ---- blend logits, per-sample softmax over views, rgb_out ----
    for (int i = tid; i < CH_ROWS; i += BLOCK) {
      float acc = FW[F_RGB_B3];
#pragma unroll
      for (int c = 0; c < 8; ++c) acc += (float)sC[i * 16 + c] * FW[F_RGB_W3 + c];
      sWgt[i] = (sMask[i] == 0.f) ? -1e9f : acc;
    }
    __syncthreads();
    for (int smp = tid; smp < CH_SAMP; smp += BLOCK) {
      float mx = -1e30f;
      for (int v = 0; v < N_VIEW; ++v) mx = fmaxf(mx, sWgt[smp * N_VIEW + v]);
      float se = 0.f, r0 = 0.f, r1 = 0.f, r2 = 0.f;
      for (int v = 0; v < N_VIEW; ++v) {
        int r = smp * N_VIEW + v;
        float p = fexp(sWgt[r] - mx);
        se += p;
        r0 += p * sRGBin[r * 3 + 0];
        r1 += p * sRGBin[r * 3 + 1];
        r2 += p * sRGBin[r * 3 + 2];
      }
      float inv = 1.f / se;
      sRGBO[(s0 + smp) * 4 + 0] = r0 * inv;
      sRGBO[(s0 + smp) * 4 + 1] = r1 * inv;
      sRGBO[(s0 + smp) * 4 + 2] = r2 * inv;
    }
    __syncthreads();
  }

  // ---- geo MLP (65->64->16), M=64 samples ----
  _Float16* sGeoT = (_Float16*)(smem + OFF_SFEAT);  // [64][64]
  _Float16* sGeo2 = (_Float16*)(smem + OFF_SX);     // [64][16]
  gemm_lds(sGFIN, 96, WH + H_GEO1T, 96, sGeoT, 64, FW + F_GEO_B1, nullptr, 0, 4, 4, 96, tid);
  __syncthreads();
  gemm_lds(sGeoT, 64, WH + H_GEO2T, 64, sGeo2, 16, FW + F_GEO_B2, nullptr, 0, 4, 1, 64, tid);
  __syncthreads();
  // inv = globalfeat + positional encoding  (10000^(-2(c/2)/16) via hw exp)
  for (int i = tid; i < 64 * 16; i += BLOCK) {
    int s = i >> 4, c = i & 15;
    float ang = (float)s * __expf(-(float)(2 * (c >> 1)) * (9.210340372f / 16.f));
    sGF[i] = (float)sGeo2[i] + ((c & 1) ? __cosf(ang) : __sinf(ang));
  }
  __syncthreads();

  float* qf = (float*)(smem + OFF_QF);
  float* kf = (float*)(smem + OFF_KF);
  float* vf = (float*)(smem + OFF_VF);
  float* sAo = (float*)(smem + OFF_AOUT);

  // ---- geometry FC on lf_tensor -> q/k tails ----
  for (int i = tid; i < 64 * 12; i += BLOCK) {
    int s = i / 12, f = i - s * 12;
    const float* lft = sLFT + s * 48 + f * 4;  // [mv][comp]
    float t1a[16], t1b[16];
    const float* w0 = FW + F_LFW0 + f * 32;
#pragma unroll
    for (int o = 0; o < 16; ++o) {
      t1a[o] = lft[0] * w0[o] + lft[2] * w0[16 + o];  // v=0 (comp 0)
      t1b[o] = lft[1] * w0[o] + lft[3] * w0[16 + o];  // v=1 (comp 1)
    }
    const float* b0 = FW + F_BETA0 + f * 16;
#pragma unroll
    for (int o = 0; o < 16; ++o) {
      float nr = sqrtf(t1a[o] * t1a[o] + t1b[o] * t1b[o]);
      float sc = eluf(nr - b0[o]) / (nr + 1e-5f);
      t1a[o] *= sc; t1b[o] *= sc;
    }
    float t2a[4] = {0.f, 0.f, 0.f, 0.f}, t2b[4] = {0.f, 0.f, 0.f, 0.f};
    const float* w1 = FW + F_LFW1 + f * 64;
#pragma unroll
    for (int o = 0; o < 16; ++o)
#pragma unroll
      for (int o2 = 0; o2 < 4; ++o2) {
        t2a[o2] += t1a[o] * w1[o * 4 + o2];
        t2b[o2] += t1b[o] * w1[o * 4 + o2];
      }
    const float* b1 = FW + F_BETA1 + f * 4;
#pragma unroll
    for (int o2 = 0; o2 < 4; ++o2) {
      float nr = sqrtf(t2a[o2] * t2a[o2] + t2b[o2] * t2b[o2]);
      float sc = eluf(nr - b1[o2]) / (nr + 1e-5f);
      t2a[o2] *= sc; t2b[o2] *= sc;
    }
#pragma unroll
    for (int h = 0; h < 4; ++h) {
      float qa = 0.f, qb = 0.f, ka = 0.f, kb = 0.f;
#pragma unroll
      for (int c = 0; c < 4; ++c) {
        qa += t2a[c] * FW[F_WQMT + f * 16 + c * 4 + h];
        qb += t2b[c] * FW[F_WQMT + f * 16 + c * 4 + h];
        ka += t2a[c] * FW[F_WKMT + f * 16 + c * 4 + h];
        kb += t2b[c] * FW[F_WKMT + f * 16 + c * 4 + h];
      }
      qf[s * 112 + h * 28 + 4 + f * 2 + 0] = qa;
      qf[s * 112 + h * 28 + 4 + f * 2 + 1] = qb;
      kf[s * 112 + h * 28 + 4 + f * 2 + 0] = ka;
      kf[s * 112 + h * 28 + 4 + f * 2 + 1] = kb;
    }
  }
  // ---- inv q/k/v heads ----
  for (int s = tid; s < 64; s += BLOCK) {
    float iv[16];
#pragma unroll
    for (int j = 0; j < 16; ++j) iv[j] = sGF[s * 16 + j];
#pragma unroll
    for (int j = 0; j < 16; ++j) {
      float q = 0.f, k = 0.f, v = 0.f;
#pragma unroll
      for (int i2 = 0; i2 < 16; ++i2) {
        q += iv[i2] * FW[F_WQI + i2 * 16 + j];
        k += iv[i2] * FW[F_WKI + i2 * 16 + j];
        v += iv[i2] * FW[F_WVI + i2 * 16 + j];
      }
      qf[s * 112 + (j >> 2) * 28 + (j & 3)] = q;
      kf[s * 112 + (j >> 2) * 28 + (j & 3)] = k;
      vf[s * 16 + j] = v;
    }
  }
  __syncthreads();
  // ---- ray attention: thread = (head, query), online softmax ----
  {
    const int h = tid >> 6, qi = tid & 63;
    float qv[28];
#pragma unroll
    for (int d = 0; d < 28; ++d) qv[d] = qf[qi * 112 + h * 28 + d] * 0.5f;  // 1/sqrt(d_k=4)
    const bool msk = (sNV[qi] <= 1.f);
    float mr = -3.0e38f, l = 0.f, o0 = 0.f, o1 = 0.f, o2 = 0.f, o3 = 0.f;
    for (int k = 0; k < 64; ++k) {
      float sc = 0.f;
      if (!msk) {
        const float* kr = kf + k * 112 + h * 28;
#pragma unroll
        for (int d = 0; d < 28; ++d) sc += qv[d] * kr[d];
      }
      float mn = fmaxf(mr, sc);
      float corr = fexp(mr - mn), p = fexp(sc - mn);
      l = l * corr + p;
      const float* vr = vf + k * 16 + h * 4;
      o0 = o0 * corr + p * vr[0];
      o1 = o1 * corr + p * vr[1];
      o2 = o2 * corr + p * vr[2];
      o3 = o3 * corr + p * vr[3];
      mr = mn;
    }
    float inv = 1.f / l;
    sAo[qi * 16 + h * 4 + 0] = o0 * inv;
    sAo[qi * 16 + h * 4 + 1] = o1 * inv;
    sAo[qi * 16 + h * 4 + 2] = o2 * inv;
    sAo[qi * 16 + h * 4 + 3] = o3 * inv;
  }
  __syncthreads();
  // ---- fc1 + residual + layernorm ----
  for (int s = tid; s < 64; s += BLOCK) {
    float o[16];
    float mu = 0.f;
#pragma unroll
    for (int j = 0; j < 16; ++j) {
      float a = sGF[s * 16 + j];
#pragma unroll
      for (int i2 = 0; i2 < 16; ++i2) a += sAo[s * 16 + i2] * FW[F_FC1 + i2 * 16 + j];
      o[j] = a;
      mu += a;
    }
    mu *= (1.f / 16.f);
    float va = 0.f;
#pragma unroll
    for (int j = 0; j < 16; ++j) { float d = o[j] - mu; va += d * d; }
    va *= (1.f / 16.f);
    float rs = rsqrtf(va + 1e-6f);
#pragma unroll
    for (int j = 0; j < 16; ++j)
      sT[s * 16 + j] = (o[j] - mu) * rs * FW[F_LNG + j] + FW[F_LNB + j];
  }
  __syncthreads();
  // ---- sigma head + output write ----
  for (int s = tid; s < 64; s += BLOCK) {
    float t[16];
#pragma unroll
    for (int j = 0; j < 16; ++j) {
      float a = FW[F_OG_B1 + j];
#pragma unroll
      for (int i2 = 0; i2 < 16; ++i2) a += sT[s * 16 + i2] * FW[F_OG_W1 + i2 * 16 + j];
      t[j] = eluf(a);
    }
    float sg = FW[F_OG_B2];
#pragma unroll
    for (int j = 0; j < 16; ++j) sg += t[j] * FW[F_OG_W2 + j];
    sg = fmaxf(sg, 0.f);
    if (sNV[s] < 1.f) sg = 0.f;
    int ob = (ray * 64 + s) * 4;
    out[ob + 0] = sRGBO[s * 4 + 0];
    out[ob + 1] = sRGBO[s * 4 + 1];
    out[ob + 2] = sRGBO[s * 4 + 2];
    out[ob + 3] = sg;
  }
}

extern "C" void kernel_launch(void* const* d_in, const int* in_sizes, int n_in,
                              void* d_out, int out_size, void* d_ws, size_t ws_size,
                              hipStream_t stream) {
  (void)in_sizes; (void)out_size; (void)ws_size;
  PP pp{};
  for (int i = 0; i < 49 && i < n_in; ++i) pp.p[i] = (const float*)d_in[i];
  _Float16* WH = (_Float16*)d_ws;
  float* FW = (float*)((char*)d_ws + WS_HALF_BYTES);
  hipFuncSetAttribute(reinterpret_cast<const void*>(ibrnet_ray_kernel),
                      hipFuncAttributeMaxDynamicSharedMemorySize, LDS_BYTES);
  ibrnet_prep_kernel<<<1, BLOCK, 0, stream>>>(pp, WH, FW);
  ibrnet_ray_kernel<<<2048, BLOCK, LDS_BYTES, stream>>>(
      pp.p[0], pp.p[1], pp.p[2], WH, FW, (float*)d_out);
}
